// EntityEncoder_87591563034961
// MI455X (gfx1250) — matrix operand order repaired and checked by reading, not compiled, not run on hardware
//
#include <hip/hip_runtime.h>
#include <hip/hip_bf16.h>
#include <math.h>

// ---------------------------------------------------------------------------
// EntityEncoder GNN for MI455X (gfx1250, wave32).
// Strategy: factor edge-level GEMMs into node/relation-level precomputes,
// do all dense GEMMs with v_wmma_f32_16x16x32_bf16 (f32 accumulate),
// memory-bound edge gather/scatter with wave-per-edge + f32 atomics (L2
// resident: agg = 51MB << 192MB L2), fused WMMA GRU (no [N,384] temps).
// ---------------------------------------------------------------------------

#define SLOPE 0.22916666666666666f   // RReLU eval slope (0.125 + 1/3)/2
#define EPS   1e-5f

typedef __attribute__((ext_vector_type(16))) __bf16 v16bf;
typedef __attribute__((ext_vector_type(8)))  float  v8f;

#define WMMA_BF16(a, b, c) \
  __builtin_amdgcn_wmma_f32_16x16x32_bf16(false, (a), false, (b), (short)0, (c), false, false)

__device__ __forceinline__ float sigmoidf_(float x) { return 1.0f / (1.0f + expf(-x)); }
__device__ __forceinline__ float rrelu_(float x)    { return x >= 0.0f ? x : SLOPE * x; }

// --------------------------- WMMA fragment loaders -------------------------
// A (16x32 bf16, MxK): lanes 0-15 row M=lane, K=0..15;
// lanes 16-31 row M=lane-16, K=16..31; VGPR v holds 2 halves.
__device__ __forceinline__ v16bf load_a_frag(const float* A, int ldA, int row0,
                                             int rows, int kk) {
  const int lane = threadIdx.x & 31;
  const int m = lane & 15, g = lane >> 4;
  int r = row0 + m;
  if (r >= rows) r = rows - 1;               // clamp; stores are row-guarded
  const float* p = A + (size_t)r * ldA + kk;
  v16bf a;
#pragma unroll
  for (int v = 0; v < 8; ++v) {
    const int kb = g * 16 + v * 2;
    a[2 * v + 0] = (__bf16)p[kb + 0];
    a[2 * v + 1] = (__bf16)p[kb + 1];
  }
  return a;
}

// B (32x16 bf16, KxN), weight stored [K, Nc] row-major ("KN").
// lanes 0-15: N=lane, K=0..15 (2/VGPR); lanes 16-31: N=lane-16, K=16..31.
__device__ __forceinline__ v16bf load_b_frag_kn(const float* W, int Nc, int col0,
                                                int kk) {
  const int lane = threadIdx.x & 31;
  const int n = lane & 15, g = lane >> 4;
  const float* p = W + (size_t)(kk + g * 16) * Nc + col0 + n;
  v16bf b;
#pragma unroll
  for (int v = 0; v < 8; ++v) {
    b[2 * v + 0] = (__bf16)p[(size_t)(2 * v + 0) * Nc];
    b[2 * v + 1] = (__bf16)p[(size_t)(2 * v + 1) * Nc];
  }
  return b;
}

// B fragment from weight stored [Nout, K] row-major ("NK", i.e. used as W^T).
__device__ __forceinline__ v16bf load_b_frag_nk(const float* W, int K, int col0,
                                                int kk) {
  const int lane = threadIdx.x & 31;
  const int n = lane & 15, g = lane >> 4;
  const float* p = W + (size_t)(col0 + n) * K + kk + g * 16;
  v16bf b;
#pragma unroll
  for (int v = 0; v < 8; ++v) {
    b[2 * v + 0] = (__bf16)p[2 * v + 0];
    b[2 * v + 1] = (__bf16)p[2 * v + 1];
  }
  return b;
}

// --------------------------- generic WMMA GEMM -----------------------------
// C[M,Nc] = act(A[M,K] @ W + bias). One wave (block of 32) per 16x16 tile.
// ACT: 0=none, 1=relu, 2=rrelu.  wlayout: 0 = W is [K,Nc], 1 = W is [Nc,K].
template <int ACT>
__global__ void gemm_wmma_kernel(const float* __restrict__ A,
                                 const float* __restrict__ W,
                                 const float* __restrict__ bias,
                                 float* __restrict__ C,
                                 int M, int K, int Nc, int wlayout) {
  const int row0 = blockIdx.x * 16;
  const int col0 = blockIdx.y * 16;
  v8f acc = {};
  for (int kk = 0; kk < K; kk += 32) {
    v16bf a = load_a_frag(A, K, row0, M, kk);
    v16bf b = wlayout ? load_b_frag_nk(W, K, col0, kk)
                      : load_b_frag_kn(W, Nc, col0, kk);
    acc = WMMA_BF16(a, b, acc);
  }
  const int lane = threadIdx.x & 31;
  const int n = lane & 15, hi = lane >> 4;
  const float bz = bias ? bias[col0 + n] : 0.0f;
#pragma unroll
  for (int v = 0; v < 8; ++v) {
    const int r = row0 + v + 8 * hi;
    if (r < M) {
      float x = acc[v] + bz;
      if (ACT == 1) x = fmaxf(x, 0.0f);
      if (ACT == 2) x = rrelu_(x);
      C[(size_t)r * Nc + col0 + n] = x;
    }
  }
}

// --------------------------- fused WMMA GRU --------------------------------
// out = (1-z)*tanh(i_n + r*h_n) + z*h ;  r=sig(i_r+h_r), z=sig(i_z+h_z)
// gi = X @ wih^T, gh = H @ whh^T ; wih/whh stored [3*128, 128] row-major.
// One wave per 16-row x 16-col output tile: 6 accumulators, 24 WMMAs.
__global__ void gru_wmma_kernel(const float* __restrict__ X,
                                const float* __restrict__ H,
                                const float* __restrict__ wih,
                                const float* __restrict__ whh,
                                const float* __restrict__ bih,
                                const float* __restrict__ bhh,
                                float* __restrict__ Out, int M) {
  const int row0 = blockIdx.x * 16;
  const int col0 = blockIdx.y * 16;
  v8f ir = {}, iz = {}, in_ = {}, hr = {}, hz = {}, hn = {};
  for (int kk = 0; kk < 128; kk += 32) {
    v16bf ax = load_a_frag(X, 128, row0, M, kk);
    v16bf ah = load_a_frag(H, 128, row0, M, kk);
    v16bf br = load_b_frag_nk(wih, 128, col0, kk);
    v16bf bz = load_b_frag_nk(wih, 128, col0 + 128, kk);
    v16bf bn = load_b_frag_nk(wih, 128, col0 + 256, kk);
    ir = WMMA_BF16(ax, br, ir);
    iz = WMMA_BF16(ax, bz, iz);
    in_ = WMMA_BF16(ax, bn, in_);
    v16bf cr = load_b_frag_nk(whh, 128, col0, kk);
    v16bf cz = load_b_frag_nk(whh, 128, col0 + 128, kk);
    v16bf cn = load_b_frag_nk(whh, 128, col0 + 256, kk);
    hr = WMMA_BF16(ah, cr, hr);
    hz = WMMA_BF16(ah, cz, hz);
    hn = WMMA_BF16(ah, cn, hn);
  }
  const int lane = threadIdx.x & 31;
  const int n = lane & 15, hi = lane >> 4;
  const float b_ir = bih[col0 + n], b_iz = bih[128 + col0 + n], b_in = bih[256 + col0 + n];
  const float b_hr = bhh[col0 + n], b_hz = bhh[128 + col0 + n], b_hn = bhh[256 + col0 + n];
#pragma unroll
  for (int v = 0; v < 8; ++v) {
    const int r = row0 + v + 8 * hi;
    if (r < M) {
      const float hval = H[(size_t)r * 128 + col0 + n];
      const float rg = sigmoidf_(ir[v] + b_ir + hr[v] + b_hr);
      const float zg = sigmoidf_(iz[v] + b_iz + hz[v] + b_hz);
      const float ng = tanhf(in_[v] + b_in + rg * (hn[v] + b_hn));
      Out[(size_t)r * 128 + col0 + n] = (1.0f - zg) * ng + zg * hval;
    }
  }
}

// ------------------------- LayerNorm (wave per row) ------------------------
__global__ void ln_kernel(const float* __restrict__ In,
                          const float* __restrict__ Res,   // optional residual
                          float* __restrict__ Out,
                          const float* __restrict__ g,
                          const float* __restrict__ b, int rows) {
  const int row = blockIdx.x;
  const int lane = threadIdx.x;
  if (row >= rows) return;
  const float* p = In + (size_t)row * 128;
  float x[4];
  float s = 0.0f;
#pragma unroll
  for (int j = 0; j < 4; ++j) {
    const int d = lane + 32 * j;
    x[j] = p[d] + (Res ? Res[(size_t)row * 128 + d] : 0.0f);
    s += x[j];
  }
#pragma unroll
  for (int off = 16; off >= 1; off >>= 1) s += __shfl_xor(s, off, 32);
  const float mu = s * (1.0f / 128.0f);
  float vv = 0.0f;
#pragma unroll
  for (int j = 0; j < 4; ++j) { const float d = x[j] - mu; vv += d * d; }
#pragma unroll
  for (int off = 16; off >= 1; off >>= 1) vv += __shfl_xor(vv, off, 32);
  const float rs = rsqrtf(vv * (1.0f / 128.0f) + EPS);
#pragma unroll
  for (int j = 0; j < 4; ++j) {
    const int d = lane + 32 * j;
    Out[(size_t)row * 128 + d] = (x[j] - mu) * rs * g[d] + b[d];
  }
}

// ---------------------- tiny query-relation precompute ---------------------
// qrW[b, j] = rel_emb[q_rel[b] + b*NR] . Wqr_w[:, j] + Wqr_b[j]   (8 x 64)
__global__ void qr_kernel(const float* __restrict__ rel_emb,
                          const int* __restrict__ q_rel,
                          const float* __restrict__ Wqr_w,
                          const float* __restrict__ Wqr_b,
                          float* __restrict__ qrW, int NR) {
  const int b = blockIdx.x;
  const int j = threadIdx.x;          // 64 threads
  const int q = q_rel[b] + b * NR;
  const float* row = rel_emb + (size_t)q * 128;
  float acc = Wqr_b[j];
  for (int k = 0; k < 128; ++k) acc += row[k] * Wqr_w[(size_t)k * 64 + j];
  qrW[b * 64 + j] = acc;
}

// ----------------------------- edge kernel ---------------------------------
// One wave per edge. alpha via wave butterfly reduction, then 128-wide
// message scatter-add into agg (f32 atomics, agg stays in L2).
__global__ void edge_kernel(const float* __restrict__ hidden,
                            const float* __restrict__ rel_emb,
                            const float* __restrict__ hsW,
                            const float* __restrict__ relW,
                            const float* __restrict__ qrW,
                            const float* __restrict__ walpha_w,
                            const float* __restrict__ walpha_b,
                            const int* __restrict__ batch_idx,
                            const int* __restrict__ rel,
                            const int* __restrict__ sub,
                            const int* __restrict__ obj,
                            float* __restrict__ agg, int E, int NR) {
  const int e = (blockIdx.x * blockDim.x + threadIdx.x) >> 5;
  const int lane = threadIdx.x & 31;
  if (e >= E) return;
  const int s = sub[e], o = obj[e], bb = batch_idx[e];
  const int rfi = rel[e] + bb * NR;
  const size_t s64 = (size_t)s * 64, r64 = (size_t)rfi * 64;
  float part = 0.0f;
#pragma unroll
  for (int j = 0; j < 2; ++j) {
    const int c = lane + 32 * j;
    const float pre = hsW[s64 + c] + relW[r64 + c] + qrW[bb * 64 + c];
    part += rrelu_(pre) * walpha_w[c];
  }
#pragma unroll
  for (int off = 16; off >= 1; off >>= 1) part += __shfl_xor(part, off, 32);
  const float alpha = sigmoidf_(part + walpha_b[0]);
  const size_t s128 = (size_t)s * 128, r128 = (size_t)rfi * 128, o128 = (size_t)o * 128;
#pragma unroll
  for (int j = 0; j < 4; ++j) {
    const int d = lane + 32 * j;
    const float m = alpha * (hidden[s128 + d] + rel_emb[r128 + d]);
    atomicAdd(&agg[o128 + d], m);
  }
}

// ----------------------------- final scatter -------------------------------
__global__ void final_kernel(const float* __restrict__ hidden,
                             const float* __restrict__ Wfinal,
                             const int* __restrict__ nb,
                             const int* __restrict__ ne,
                             float* __restrict__ out, int Nn, int NENT) {
  const int n = (blockIdx.x * blockDim.x + threadIdx.x) >> 5;
  const int lane = threadIdx.x & 31;
  if (n >= Nn) return;
  float s = 0.0f;
#pragma unroll
  for (int j = 0; j < 4; ++j) {
    const int d = lane + 32 * j;
    s += hidden[(size_t)n * 128 + d] * Wfinal[d];
  }
#pragma unroll
  for (int off = 16; off >= 1; off >>= 1) s += __shfl_xor(s, off, 32);
  if (lane == 0) out[(size_t)nb[n] * NENT + ne[n]] = s;
}

// ------------------------------- utilities ---------------------------------
__global__ void fillf_kernel(float* __restrict__ p, float v, size_t n) {
  for (size_t i = (size_t)blockIdx.x * blockDim.x + threadIdx.x; i < n;
       i += (size_t)gridDim.x * blockDim.x)
    p[i] = v;
}
__global__ void copyf_kernel(float* __restrict__ dst, const float* __restrict__ src,
                             size_t n) {
  for (size_t i = (size_t)blockIdx.x * blockDim.x + threadIdx.x; i < n;
       i += (size_t)gridDim.x * blockDim.x)
    dst[i] = src[i];
}

// ------------------------------- host driver -------------------------------
extern "C" void kernel_launch(void* const* d_in, const int* in_sizes, int n_in,
                              void* d_out, int out_size, void* d_ws, size_t ws_size,
                              hipStream_t stream) {
  const int D = 128, A = 64, L = 2;
  const int N    = in_sizes[0] / D;        // 100000
  const int NRL  = in_sizes[1] / D;        // 1608 = B*(R+1)
  const int E    = in_sizes[2];            // 800000
  const int Bb   = in_sizes[6];            // 8
  const int NR   = NRL / Bb;               // 201
  const int NENT = 50000;

  const float* in_hidden = (const float*)d_in[0];
  const float* in_rel    = (const float*)d_in[1];
  const int*   batch_idx = (const int*)d_in[2];
  const int*   rel       = (const int*)d_in[3];
  const int*   sub       = (const int*)d_in[4];
  const int*   obj       = (const int*)d_in[5];
  const int*   q_rel     = (const int*)d_in[6];
  const int*   nodes_b   = (const int*)d_in[7];
  const int*   nodes_e   = (const int*)d_in[8];
  const float* Ws        = (const float*)d_in[9];    // [L,128,64]
  const float* Wr        = (const float*)d_in[10];   // [L,128,64]
  const float* Wqr_w     = (const float*)d_in[11];   // [L,128,64]
  const float* Wqr_b     = (const float*)d_in[12];   // [L,64]
  const float* Walpha_w  = (const float*)d_in[13];   // [L,64]
  const float* Walpha_b  = (const float*)d_in[14];   // [L]
  const float* Wh        = (const float*)d_in[15];   // [L,128,128]
  const float* rt_w      = (const float*)d_in[16];   // [L,128,128]
  const float* rt_b      = (const float*)d_in[17];   // [L,128]
  const float* ln_g      = (const float*)d_in[18];
  const float* ln_b      = (const float*)d_in[19];
  const float* lnr_g     = (const float*)d_in[20];
  const float* lnr_b     = (const float*)d_in[21];
  const float* gru_wih   = (const float*)d_in[22];   // [384,128]
  const float* gru_whh   = (const float*)d_in[23];
  const float* gru_bih   = (const float*)d_in[24];
  const float* gru_bhh   = (const float*)d_in[25];
  const float* Wfinal    = (const float*)d_in[26];
  float* out = (float*)d_out;

  // workspace carve-up (~232 MB of f32)
  float* w = (float*)d_ws;
  float* REL    = w; w += (size_t)NRL * D;
  float* RELTMP = w; w += (size_t)NRL * D;
  float* RELW   = w; w += (size_t)NRL * A;
  float* QRW    = w; w += 512;
  float* HSW    = w; w += (size_t)N * A;
  float* AGG    = w; w += (size_t)N * D;
  float* X      = w; w += (size_t)N * D;
  float* HA     = w; w += (size_t)N * D;
  float* HB     = w; w += (size_t)N * D;

  const size_t ND = (size_t)N * D;
  const dim3 blk32(32), blk64(64), blk256(256);
  const int GS = 2048;  // grid-stride blocks for fill/copy

  // ---- init: HA = hidden, HB = 0, REL = rel_emb
  copyf_kernel<<<GS, blk256, 0, stream>>>(HA, in_hidden, ND);
  fillf_kernel<<<GS, blk256, 0, stream>>>(HB, 0.0f, ND);
  copyf_kernel<<<GS, blk256, 0, stream>>>(REL, in_rel, (size_t)NRL * D);

  const int mtN = (N + 15) / 16;      // 6250
  const int mtR = (NRL + 15) / 16;    // 101

  float* hid   = HA;   // current hidden
  float* hprev = HB;   // GRU h0 (zeros for layer 0)

  for (int i = 0; i < L; ++i) {
    const float* rt_wi = rt_w + (size_t)i * D * D;
    const float* Wsi   = Ws   + (size_t)i * D * A;
    const float* Wri   = Wr   + (size_t)i * D * A;
    const float* Whi   = Wh   + (size_t)i * D * D;

    // rel transfer: RELTMP = relu(REL @ rt_w + rt_b); REL = LN(REL + RELTMP)
    gemm_wmma_kernel<1><<<dim3(mtR, D / 16), blk32, 0, stream>>>(
        REL, rt_wi, rt_b + (size_t)i * D, RELTMP, NRL, D, D, 0);
    ln_kernel<<<NRL, blk32, 0, stream>>>(REL, RELTMP, REL,
                                         lnr_g + (size_t)i * D, lnr_b + (size_t)i * D, NRL);

    // precomputes for the edge phase
    gemm_wmma_kernel<0><<<dim3(mtN, A / 16), blk32, 0, stream>>>(
        hid, Wsi, nullptr, HSW, N, D, A, 0);
    gemm_wmma_kernel<0><<<dim3(mtR, A / 16), blk32, 0, stream>>>(
        REL, Wri, nullptr, RELW, NRL, D, A, 0);
    qr_kernel<<<Bb, blk64, 0, stream>>>(REL, q_rel,
                                        Wqr_w + (size_t)i * D * A,
                                        Wqr_b + (size_t)i * A, QRW, NR);

    // edge gather -> alpha -> scatter-add
    float* aggbuf = (i == 0) ? AGG : HB;    // layer1: HB is free, AGG is h0
    fillf_kernel<<<GS, blk256, 0, stream>>>(aggbuf, 0.0f, ND);
    edge_kernel<<<(E * 32 + 255) / 256, blk256, 0, stream>>>(
        hid, REL, HSW, RELW, QRW, Walpha_w + (size_t)i * A, Walpha_b + i,
        batch_idx, rel, sub, obj, aggbuf, E, NR);

    // X = LN(rrelu(agg @ Wh))
    gemm_wmma_kernel<2><<<dim3(mtN, D / 16), blk32, 0, stream>>>(
        aggbuf, Whi, nullptr, X, N, D, D, 0);
    ln_kernel<<<N, blk32, 0, stream>>>(X, nullptr, X,
                                       ln_g + (size_t)i * D, ln_b + (size_t)i * D, N);

    // fused GRU: hidden = GRU(X, hprev) -> free buffer
    float* gout = (i == 0) ? AGG : HA;      // layer0: AGG consumed; layer1: HA free
    gru_wmma_kernel<<<dim3(mtN, D / 16), blk32, 0, stream>>>(
        X, hprev, gru_wih, gru_whh, gru_bih, gru_bhh, gout, N);
    hid = gout;
    hprev = gout;                            // h0 = hidden for next layer

    // REL = LN(REL)
    ln_kernel<<<NRL, blk32, 0, stream>>>(REL, nullptr, REL,
                                         lnr_g + (size_t)i * D, lnr_b + (size_t)i * D, NRL);
  }

  // final scores -> scatter into [B, NENT]
  fillf_kernel<<<GS, blk256, 0, stream>>>(out, 0.0f, (size_t)out_size);
  final_kernel<<<(N * 32 + 255) / 256, blk256, 0, stream>>>(
      hid, Wfinal, nodes_b, nodes_e, out, N, NENT);
}
